// TemporalAttentionLayer_16114717294629
// MI455X (gfx1250) — compile-verified, matrix operand order checked by reading, not run on hardware
//
#include <hip/hip_runtime.h>
#include <hip/hip_bf16.h>

// ---------------------------------------------------------------------------
// CDNA5 (gfx1250) fused relative-position attention layer.
// f16 WMMA (v_wmma_f32_16x16x32_f16) for all GEMMs, f32 accumulate.
// Flash-attention style streaming: the [B,H,S,S] score tensor never hits HBM.
// K/V tiles are staged in double-buffered LDS via GLOBAL_LOAD_ASYNC_TO_LDS
// (ASYNCcnt) and shared by all 4 waves of the block (4x L2 traffic cut).
// ---------------------------------------------------------------------------

#define S_LEN   1024
#define DMODEL  1024
#define NHEADS  16
#define DK      64
#define BATCH   8
#define NTOK    (BATCH * S_LEN)   // 8192 tokens
#define NBH     (BATCH * NHEADS)  // 128
#define TREL    2048              // padded clipped relative table length

// Flip to 0 if the async inline-asm ever fails to assemble: falls back to
// plain global_load_b128 + ds_store_b128 staging (same double buffering).
#define USE_ASYNC_COPY 1

typedef __attribute__((ext_vector_type(16))) _Float16 v16h;
typedef __attribute__((ext_vector_type(8)))  float    v8f;

union Frag {
  v16h h;
  uint4 q[2];
};

__device__ __forceinline__ v8f wmma16(v16h a, v16h b, v8f c) {
  // D = A(16x32 f16) * B(32x16 f16) + C(16x16 f32)
  return __builtin_amdgcn_wmma_f32_16x16x32_f16(false, a, false, b, (short)0, c,
                                                false, false);
}

__device__ __forceinline__ v8f zero8() {
  v8f r;
#pragma unroll
  for (int i = 0; i < 8; ++i) r[i] = 0.0f;
  return r;
}

// Two aligned 16B chunks -> one 16-element f16 fragment register.
__device__ __forceinline__ Frag ld_frag(const _Float16* p) {
  Frag f;
  f.q[0] = *(const uint4*)(p);
  f.q[1] = *(const uint4*)(p + 16);
  return f;
}

// 16B global -> LDS copy. Async variant issues GLOBAL_LOAD_ASYNC_TO_LDS_B128
// (VDST = 32-bit LDS byte address, VADDR = 64-bit global address), tracked by
// ASYNCcnt; generic LDS pointers carry the LDS byte offset in their low 32b.
__device__ __forceinline__ void stage16(_Float16* lp, const _Float16* gp) {
#if USE_ASYNC_COPY
  asm volatile("global_load_async_to_lds_b128 %0, %1, off"
               :
               : "v"((unsigned)(unsigned long long)(uintptr_t)lp),
                 "v"((unsigned long long)(uintptr_t)gp)
               : "memory");
#else
  *(uint4*)lp = *(const uint4*)gp;
#endif
}

__device__ __forceinline__ void wait_async_le4() {
#if USE_ASYNC_COPY
  asm volatile("s_wait_asynccnt 0x4" ::: "memory");
#endif
}
__device__ __forceinline__ void wait_async_le0() {
#if USE_ASYNC_COPY
  asm volatile("s_wait_asynccnt 0x0" ::: "memory");
#endif
}

// ---------------------------------------------------------------------------
// Kernel 1: precision conversion + clip-padded relative embedding tables.
//   Ek [t][k]  with t = (j-i)+1023 in [0,2047], clip folded into the table.
//   EvT[d][t]  transposed so attn*rel_v B-fragments are contiguous over t.
// ---------------------------------------------------------------------------
__global__ void prep_kernel(const float* __restrict__ x,
                            const float* __restrict__ wq, const float* __restrict__ wk,
                            const float* __restrict__ wv, const float* __restrict__ wo,
                            const float* __restrict__ relk, const float* __restrict__ relv,
                            _Float16* __restrict__ Xh,
                            _Float16* __restrict__ Wqh, _Float16* __restrict__ Wkh,
                            _Float16* __restrict__ Wvh, _Float16* __restrict__ Woh,
                            _Float16* __restrict__ Ek, _Float16* __restrict__ EvT) {
  const int stride = gridDim.x * blockDim.x;
  const int t0 = blockIdx.x * blockDim.x + threadIdx.x;
  for (int i = t0; i < NTOK * DMODEL; i += stride) Xh[i] = (_Float16)x[i];
  for (int i = t0; i < DMODEL * DMODEL; i += stride) {
    Wqh[i] = (_Float16)wq[i];
    Wkh[i] = (_Float16)wk[i];
    Wvh[i] = (_Float16)wv[i];
    Woh[i] = (_Float16)wo[i];
  }
  for (int i = t0; i < TREL * DK; i += stride) {
    const int t = i >> 6, k = i & 63;
    int r = t - 1023;
    r = r < -512 ? -512 : (r > 512 ? 512 : r);
    r += 512;
    Ek[i] = (_Float16)relk[r * DK + k];
  }
  for (int i = t0; i < DK * TREL; i += stride) {
    const int d = i >> 11, t = i & 2047;
    int r = t - 1023;
    r = r < -512 ? -512 : (r > 512 ? 512 : r);
    r += 512;
    EvT[i] = (_Float16)relv[r * DK + d];
  }
}

// ---------------------------------------------------------------------------
// Kernel 2: Q/K/V projection  out[n,d] = sum_e X[n,e] * W[d,e]  (x @ W^T).
// One wave computes a 16x64 output tile (A fragment reused over 4 N-tiles).
// Q,K stored [bh][s][dk]; V stored transposed [bh][dk][s].
// ---------------------------------------------------------------------------
__global__ void proj_kernel(const _Float16* __restrict__ Xh,
                            const _Float16* __restrict__ Wq,
                            const _Float16* __restrict__ Wk,
                            const _Float16* __restrict__ Wv,
                            _Float16* __restrict__ Qw,
                            _Float16* __restrict__ Kw,
                            _Float16* __restrict__ Vtw) {
  const int lane = threadIdx.x & 31;
  const int n16 = lane & 15;   // A row / B col / C col
  const int g   = lane >> 4;   // lane half-group
  const int r0  = blockIdx.x * 16;   // token tile
  const int c0  = blockIdx.y * 64;   // dmodel column group
  const int mat = blockIdx.z;        // 0=Q 1=K 2=V
  const _Float16* W = (mat == 0) ? Wq : (mat == 1) ? Wk : Wv;

  v8f acc[4];
#pragma unroll
  for (int nt = 0; nt < 4; ++nt) acc[nt] = zero8();

  for (int k0 = 0; k0 < DMODEL; k0 += 32) {
    const _Float16* ap = Xh + (size_t)(r0 + n16) * DMODEL + k0 + g * 8;
    __builtin_prefetch(ap + 64, 0, 1);
    const Frag a = ld_frag(ap);
#pragma unroll
    for (int nt = 0; nt < 4; ++nt) {
      const _Float16* bp = W + (size_t)(c0 + nt * 16 + n16) * DMODEL + k0 + g * 8;
      const Frag b = ld_frag(bp);
      acc[nt] = wmma16(a.h, b.h, acc[nt]);
    }
  }

#pragma unroll
  for (int nt = 0; nt < 4; ++nt) {
    const int dcol = c0 + nt * 16 + n16;
    const int h = dcol >> 6, d = dcol & 63;
#pragma unroll
    for (int v = 0; v < 8; ++v) {
      const int tok = r0 + v + 8 * g;
      const int b = tok >> 10, s = tok & 1023;
      const int bh = b * NHEADS + h;
      const _Float16 val = (_Float16)acc[nt][v];
      if (mat == 0)      Qw[((size_t)bh * S_LEN + s) * DK + d] = val;
      else if (mat == 1) Kw[((size_t)bh * S_LEN + s) * DK + d] = val;
      else               Vtw[((size_t)bh * DK + d) * S_LEN + s] = val;
    }
  }
}

// ---------------------------------------------------------------------------
// Kernel 3: fused flash attention with relative position bias (K and V side).
// One wave = 16 query rows of one (b,h); 4 waves/block share async-staged
// K/V tiles (32 keys) in double-buffered LDS.
//   scores = QK^T/8 + skew-gather( Q @ Ek[window]^T )
//   acc   += P @ V  +  skew-scatter(P) @ Ev[window]
// ---------------------------------------------------------------------------
#define ATT_WAVES 4

__global__ void attn_kernel(const _Float16* __restrict__ Qw,
                            const _Float16* __restrict__ Kw,
                            const _Float16* __restrict__ Vtw,
                            const _Float16* __restrict__ Ek,
                            const _Float16* __restrict__ EvT,
                            _Float16* __restrict__ Ctx) {
  __shared__ __attribute__((aligned(16))) float    ldsT[ATT_WAVES][16 * 32];
  __shared__ __attribute__((aligned(16))) _Float16 ldsP[ATT_WAVES][16 * 32];
  __shared__ __attribute__((aligned(16))) _Float16 ldsS[ATT_WAVES][16 * 32];
  __shared__ __attribute__((aligned(16))) _Float16 ldsK[2][32 * DK];  // [j][dk]
  __shared__ __attribute__((aligned(16))) _Float16 ldsV[2][DK * 32];  // [dk][j]

  const int tid  = threadIdx.x;
  const int wave = tid >> 5;
  const int lane = tid & 31;
  const int n16 = lane & 15;
  const int g   = lane >> 4;
  const int i0  = (blockIdx.x * ATT_WAVES + wave) * 16;  // query tile start
  const int bh  = blockIdx.y;

  const _Float16* Qp = Qw + (size_t)bh * S_LEN * DK;
  const _Float16* Kp = Kw + (size_t)bh * S_LEN * DK;
  const _Float16* Vp = Vtw + (size_t)bh * DK * S_LEN;

  // Cooperative stage of one 32-key K/V tile (128 threads x 4 x 16B = 8KB).
  auto stage_tile = [&](int buf, int jn) {
#pragma unroll
    for (int it = 0; it < 2; ++it) {
      const int idx = tid + it * 128;
      const int kr = idx >> 3, kc = idx & 7;  // K: 32 rows x 8 16B-chunks
      stage16(&ldsK[buf][kr * DK + kc * 8],
              Kp + (size_t)(jn + kr) * DK + kc * 8);
      const int vr = idx >> 2, vc = idx & 3;  // V: 64 rows x 4 16B-chunks
      stage16(&ldsV[buf][vr * 32 + vc * 8],
              Vp + (size_t)vr * S_LEN + jn + vc * 8);
    }
  };

  // Q A-fragments (16 rows x dk=64, two k-chunks of 32), kept in registers.
  Frag aQ[2];
#pragma unroll
  for (int c = 0; c < 2; ++c)
    aQ[c] = ld_frag(Qp + (size_t)(i0 + n16) * DK + c * 32 + g * 8);

  v8f acc[4];  // O tile: 16 rows x dk=64
#pragma unroll
  for (int nt = 0; nt < 4; ++nt) acc[nt] = zero8();
  float mrow[8], lrow[8];
#pragma unroll
  for (int v = 0; v < 8; ++v) { mrow[v] = -3.0e38f; lrow[v] = 0.0f; }

  stage_tile(0, 0);  // prologue: async-fill buffer 0

  for (int j0 = 0; j0 < S_LEN; j0 += 32) {
    const int buf = (j0 >> 5) & 1;
    const bool more = (j0 + 32) < S_LEN;
    __syncthreads();                       // buf^1 free to overwrite
    if (more) stage_tile(buf ^ 1, j0 + 32);
    if (more) wait_async_le4();            // drain older batch -> buf ready
    else      wait_async_le0();
    __syncthreads();                       // all waves' staging visible

    float sv[2][8];
#pragma unroll
    for (int jt = 0; jt < 2; ++jt) {
      const int js = j0 + jt * 16;
      const int tb = js - i0 + 1008;  // multiple of 16 -> aligned table loads
      // --- Q K^T (K from LDS tile) ---
      v8f sc = zero8();
#pragma unroll
      for (int c = 0; c < 2; ++c) {
        const Frag bK =
            ld_frag(&ldsK[buf][(jt * 16 + n16) * DK + c * 32 + g * 8]);
        sc = wmma16(aQ[c].h, bK.h, sc);
      }
      // --- rel_k: T[ii,o] = Q . Ek[tb+o], o in [0,32) ---
#pragma unroll
      for (int ot = 0; ot < 2; ++ot) {
        v8f T = zero8();
#pragma unroll
        for (int c = 0; c < 2; ++c) {
          const Frag bE =
              ld_frag(Ek + (size_t)(tb + ot * 16 + n16) * DK + c * 32 + g * 8);
          T = wmma16(aQ[c].h, bE.h, T);
        }
#pragma unroll
        for (int v = 0; v < 8; ++v)
          ldsT[wave][(v + 8 * g) * 32 + ot * 16 + n16] = T[v];
      }
      // skew-gather: scores[m,n] = qk/8 + T[m][n-m+15]
#pragma unroll
      for (int v = 0; v < 8; ++v) {
        const int m = v + 8 * g;
        sv[jt][v] = sc[v] * 0.125f + ldsT[wave][m * 32 + (n16 - m + 15)];
      }
    }

    // --- online softmax over the 32 new columns (joint max) ---
    float al[8];
#pragma unroll
    for (int v = 0; v < 8; ++v) {
      float rm = fmaxf(sv[0][v], sv[1][v]);
#pragma unroll
      for (int off = 8; off >= 1; off >>= 1)
        rm = fmaxf(rm, __shfl_xor(rm, off, 32));
      const float mnew = fmaxf(mrow[v], rm);
      al[v] = __expf(mrow[v] - mnew);
      const float p0 = __expf(sv[0][v] - mnew);
      const float p1 = __expf(sv[1][v] - mnew);
      float rs = p0 + p1;
#pragma unroll
      for (int off = 8; off >= 1; off >>= 1) rs += __shfl_xor(rs, off, 32);
      lrow[v] = lrow[v] * al[v] + rs;
      mrow[v] = mnew;
      sv[0][v] = p0;
      sv[1][v] = p1;
    }
#pragma unroll
    for (int nt = 0; nt < 4; ++nt)
#pragma unroll
      for (int v = 0; v < 8; ++v) acc[nt][v] *= al[v];

    // probabilities -> LDS (natural layout for P@V)
#pragma unroll
    for (int jt = 0; jt < 2; ++jt)
#pragma unroll
      for (int v = 0; v < 8; ++v) {
        const int m = v + 8 * g;
        ldsP[wave][m * 32 + jt * 16 + n16] = (_Float16)sv[jt][v];
      }

    // --- rel_v: skew-scatter P then GEMM against Ev window ---
#pragma unroll
    for (int jt = 0; jt < 2; ++jt) {
      const int tb = j0 + jt * 16 - i0 + 1008;
#pragma unroll
      for (int e = 0; e < 16; ++e) ldsS[wave][lane * 16 + e] = (_Float16)0.0f;
#pragma unroll
      for (int v = 0; v < 8; ++v) {
        const int m = v + 8 * g;
        ldsS[wave][m * 32 + (n16 - m + 15)] = (_Float16)sv[jt][v];
      }
      const Frag aS = ld_frag(&ldsS[wave][n16 * 32 + g * 8]);
#pragma unroll
      for (int nt = 0; nt < 4; ++nt) {
        const Frag bE =
            ld_frag(EvT + (size_t)(nt * 16 + n16) * TREL + tb + g * 8);
        acc[nt] = wmma16(aS.h, bE.h, acc[nt]);
      }
    }

    // --- P @ V over all 32 columns (V from LDS tile) ---
    const Frag aP = ld_frag(&ldsP[wave][n16 * 32 + g * 8]);
#pragma unroll
    for (int nt = 0; nt < 4; ++nt) {
      const Frag bV = ld_frag(&ldsV[buf][(nt * 16 + n16) * 32 + g * 8]);
      acc[nt] = wmma16(aP.h, bV.h, acc[nt]);
    }
  }

  // epilogue: normalize by l, write ctx as [tok][dmodel] f16
  const int b = bh >> 4, h = bh & 15;
  float inv[8];
#pragma unroll
  for (int v = 0; v < 8; ++v) inv[v] = 1.0f / lrow[v];
#pragma unroll
  for (int nt = 0; nt < 4; ++nt)
#pragma unroll
    for (int v = 0; v < 8; ++v) {
      const int tok = i0 + v + 8 * g;
      const int col = h * DK + nt * 16 + n16;
      Ctx[((size_t)(b * S_LEN + tok)) * DMODEL + col] =
          (_Float16)(acc[nt][v] * inv[v]);
    }
}

// ---------------------------------------------------------------------------
// Kernel 4: out = LN( ctx @ Wo^T + b_o + x ).  16 rows per block (8 waves),
// ctx rows staged in LDS, row stats via ds_add_f32 LDS atomics.
// ---------------------------------------------------------------------------
__global__ void outln_kernel(const _Float16* __restrict__ Ctx,
                             const _Float16* __restrict__ Wo,
                             const float* __restrict__ b_o,
                             const float* __restrict__ x,
                             const float* __restrict__ gamma,
                             const float* __restrict__ beta,
                             float* __restrict__ out) {
  __shared__ __attribute__((aligned(16))) _Float16 ldsC[16 * DMODEL];
  __shared__ float ldsSum[16];
  __shared__ float ldsSq[16];

  const int tid  = threadIdx.x;
  const int wave = tid >> 5;
  const int lane = tid & 31;
  const int n16 = lane & 15;
  const int g   = lane >> 4;
  const int r0  = blockIdx.x * 16;

  // stage 16 ctx rows (32KB) into LDS
  {
    const uint4* src = (const uint4*)(Ctx + (size_t)r0 * DMODEL);
    uint4* dst = (uint4*)ldsC;
    for (int i = tid; i < 16 * DMODEL / 8; i += 256) dst[i] = src[i];
  }
  if (tid < 16) { ldsSum[tid] = 0.0f; ldsSq[tid] = 0.0f; }
  __syncthreads();

  v8f cAcc[8];
#pragma unroll
  for (int nt = 0; nt < 8; ++nt) cAcc[nt] = zero8();

  for (int k0 = 0; k0 < DMODEL; k0 += 32) {
    const Frag a = ld_frag(&ldsC[n16 * DMODEL + k0 + g * 8]);
#pragma unroll
    for (int nt = 0; nt < 8; ++nt) {
      const _Float16* wp =
          Wo + (size_t)(wave * 128 + nt * 16 + n16) * DMODEL + k0 + g * 8;
      __builtin_prefetch(wp + 64, 0, 1);
      const Frag bW = ld_frag(wp);
      cAcc[nt] = wmma16(a.h, bW.h, cAcc[nt]);
    }
  }

  // bias + residual; accumulate row stats
  float psum[8], psq[8];
#pragma unroll
  for (int v = 0; v < 8; ++v) { psum[v] = 0.0f; psq[v] = 0.0f; }
#pragma unroll
  for (int nt = 0; nt < 8; ++nt)
#pragma unroll
    for (int v = 0; v < 8; ++v) {
      const int m = v + 8 * g;
      const int col = wave * 128 + nt * 16 + n16;
      const float val = cAcc[nt][v] + b_o[col] + x[(size_t)(r0 + m) * DMODEL + col];
      cAcc[nt][v] = val;
      psum[v] += val;
      psq[v] += val * val;
    }
#pragma unroll
  for (int v = 0; v < 8; ++v) {
#pragma unroll
    for (int off = 8; off >= 1; off >>= 1) {
      psum[v] += __shfl_xor(psum[v], off, 32);
      psq[v]  += __shfl_xor(psq[v], off, 32);
    }
    if (n16 == 0) {
      atomicAdd(&ldsSum[v + 8 * g], psum[v]);
      atomicAdd(&ldsSq[v + 8 * g], psq[v]);
    }
  }
  __syncthreads();

#pragma unroll
  for (int nt = 0; nt < 8; ++nt)
#pragma unroll
    for (int v = 0; v < 8; ++v) {
      const int m = v + 8 * g;
      const int col = wave * 128 + nt * 16 + n16;
      const float mu  = ldsSum[m] * (1.0f / DMODEL);
      const float var = ldsSq[m] * (1.0f / DMODEL) - mu * mu;
      const float rs = rsqrtf(var + 1e-5f);
      out[(size_t)(r0 + m) * DMODEL + col] =
          (cAcc[nt][v] - mu) * rs * gamma[col] + beta[col];
    }
}

// ---------------------------------------------------------------------------
// Launcher.  Workspace layout (f16, ~92 MB total):
//   Xh | Wq Wk Wv Wo | Ek EvT | Q K Vt | ctx
// ---------------------------------------------------------------------------
extern "C" void kernel_launch(void* const* d_in, const int* in_sizes, int n_in,
                              void* d_out, int out_size, void* d_ws, size_t ws_size,
                              hipStream_t stream) {
  const float* x     = (const float*)d_in[0];
  const float* w_q   = (const float*)d_in[1];
  const float* w_k   = (const float*)d_in[2];
  const float* w_v   = (const float*)d_in[3];
  const float* w_o   = (const float*)d_in[4];
  const float* b_o   = (const float*)d_in[5];
  const float* relk  = (const float*)d_in[6];
  const float* relv  = (const float*)d_in[7];
  const float* gamma = (const float*)d_in[8];
  const float* beta  = (const float*)d_in[9];
  float* out = (float*)d_out;

  char* wp = (char*)d_ws;
  _Float16* Xh  = (_Float16*)wp; wp += (size_t)NTOK * DMODEL * 2;
  _Float16* Wqh = (_Float16*)wp; wp += (size_t)DMODEL * DMODEL * 2;
  _Float16* Wkh = (_Float16*)wp; wp += (size_t)DMODEL * DMODEL * 2;
  _Float16* Wvh = (_Float16*)wp; wp += (size_t)DMODEL * DMODEL * 2;
  _Float16* Woh = (_Float16*)wp; wp += (size_t)DMODEL * DMODEL * 2;
  _Float16* Ek  = (_Float16*)wp; wp += (size_t)TREL * DK * 2;
  _Float16* EvT = (_Float16*)wp; wp += (size_t)DK * TREL * 2;
  _Float16* Qw  = (_Float16*)wp; wp += (size_t)NBH * S_LEN * DK * 2;
  _Float16* Kw  = (_Float16*)wp; wp += (size_t)NBH * S_LEN * DK * 2;
  _Float16* Vtw = (_Float16*)wp; wp += (size_t)NBH * DK * S_LEN * 2;
  _Float16* Ctx = (_Float16*)wp; wp += (size_t)NTOK * DMODEL * 2;

  prep_kernel<<<2048, 256, 0, stream>>>(x, w_q, w_k, w_v, w_o, relk, relv,
                                        Xh, Wqh, Wkh, Wvh, Woh, Ek, EvT);

  proj_kernel<<<dim3(NTOK / 16, DMODEL / 64, 3), dim3(32), 0, stream>>>(
      Xh, Wqh, Wkh, Wvh, Qw, Kw, Vtw);

  attn_kernel<<<dim3((S_LEN / 16) / ATT_WAVES, NBH), dim3(32 * ATT_WAVES), 0,
                stream>>>(Qw, Kw, Vtw, Ek, EvT, Ctx);

  outln_kernel<<<dim3(NTOK / 16), dim3(256), 0, stream>>>(Ctx, Woh, b_o, x,
                                                          gamma, beta, out);
}
